// MGAttention_86947317940375
// MI455X (gfx1250) — matrix operand order repaired and checked by reading, not compiled, not run on hardware
//
#include <hip/hip_runtime.h>

// ---------------------------------------------------------------------------
// MGAttention on MI455X (gfx1250): bf16 WMMA pipeline.
//   X(8192x768) -> QKV proj -> 64 independent (1024x64) attention blocks
//   (flash-style online softmax) -> out proj (8192x768 fp32).
// All matmuls via v_wmma_f32_16x16x32_bf16.
// ---------------------------------------------------------------------------

typedef __attribute__((ext_vector_type(16))) __bf16 v16bf;
typedef __attribute__((ext_vector_type(8)))  float  v8f;
typedef __attribute__((ext_vector_type(4)))  unsigned int v4u;

union FragB16 {
  v16bf v;
  v4u   q[2];
};

#define SEQ    8192   // B*N flattened rows
#define DMODEL 768
#define ADIM   512
#define NG     64     // number of (h,b) attention blocks
#define GROWS  1024   // rows per block after reshape
#define DH     64
#define SCALE  0.35355339059327373f  // 8^-0.5

__device__ __forceinline__ unsigned short f2bf(float f) {
  unsigned int u = __float_as_uint(f);
  u += 0x7FFFu + ((u >> 16) & 1u);   // round-to-nearest-even
  return (unsigned short)(u >> 16);
}

__device__ __forceinline__ v8f wmma_bf16(const FragB16& a, const FragB16& b, v8f c) {
  return __builtin_amdgcn_wmma_f32_16x16x32_bf16(
      /*neg_a=*/false, a.v, /*neg_b=*/false, b.v,
      /*c_mod=*/(short)0, c, /*reuse_a=*/false, /*reuse_b=*/false);
}

// ------------------------------ prep kernels -------------------------------

__global__ __launch_bounds__(256) void cvt_f32_to_bf16(
    const float* __restrict__ src, unsigned short* __restrict__ dst, int n) {
  int i = blockIdx.x * blockDim.x + threadIdx.x;
  if (i < n) dst[i] = f2bf(src[i]);
}

// dst[n*K + k] = bf16(src[k*N + n]); src is K x N row-major f32.
__global__ __launch_bounds__(256) void transpose_to_bf16(
    const float* __restrict__ src, unsigned short* __restrict__ dst, int K, int N) {
  int i = blockIdx.x * blockDim.x + threadIdx.x;
  if (i < K * N) {
    int n = i / K, k = i % K;
    dst[i] = f2bf(src[k * N + n]);
  }
}

// ------------------------- fused QKV projection ----------------------------
// C(8192x512) = Xbf(8192x768) @ W + b, one GEMM per blockIdx.z (Q/K/V).
// Block: 256 thr = 8 waves, tile 128x128. Wave = 64 rows x 32 cols
// (4 M-tiles x 2 N-tiles = 8 WMMA per 32-wide K step).
__global__ __launch_bounds__(256) void qkv_gemm(
    const unsigned short* __restrict__ X,
    const unsigned short* __restrict__ Wqt,
    const unsigned short* __restrict__ Wkt,
    const unsigned short* __restrict__ Wvt,
    const float* __restrict__ bq, const float* __restrict__ bk,
    const float* __restrict__ bv,
    unsigned short* __restrict__ Qo, unsigned short* __restrict__ Ko,
    unsigned short* __restrict__ Vt) {
  const int which = blockIdx.z;
  const unsigned short* Wt = (which == 0) ? Wqt : (which == 1) ? Wkt : Wvt;
  const float* bias = (which == 0) ? bq : (which == 1) ? bk : bv;

  const int m0   = blockIdx.y * 128;
  const int n0   = blockIdx.x * 128;
  const int wave = threadIdx.x >> 5;
  const int lane = threadIdx.x & 31;
  const int lo = lane & 15, hi = lane >> 4;
  const int rowBase = m0 + (wave >> 2) * 64;
  const int colBase = n0 + (wave & 3) * 32;

  v8f acc[4][2];
#pragma unroll
  for (int i = 0; i < 4; ++i)
#pragma unroll
    for (int j = 0; j < 2; ++j)
#pragma unroll
      for (int e = 0; e < 8; ++e) acc[i][j][e] = 0.0f;

  for (int k0 = 0; k0 < DMODEL; k0 += 32) {
    FragB16 a[4], b[2];
#pragma unroll
    for (int mt = 0; mt < 4; ++mt) {
      // A layout: lane lo = row M; hi selects 16-wide K half; K contiguous.
      const unsigned short* ap = X + (rowBase + mt * 16 + lo) * DMODEL + k0;
      a[mt].q[0] = *(const v4u*)(ap + hi * 16);
      a[mt].q[1] = *(const v4u*)(ap + hi * 16 + 8);
    }
#pragma unroll
    for (int nt = 0; nt < 2; ++nt) {
      // B layout: lane lo = column N; hi selects 16-wide K half; K contiguous.
      const unsigned short* bp =
          Wt + (colBase + nt * 16 + lo) * DMODEL + k0 + hi * 16;
      b[nt].q[0] = *(const v4u*)bp;
      b[nt].q[1] = *(const v4u*)(bp + 8);
    }
#pragma unroll
    for (int mt = 0; mt < 4; ++mt)
#pragma unroll
      for (int nt = 0; nt < 2; ++nt)
        acc[mt][nt] = wmma_bf16(a[mt], b[nt], acc[mt][nt]);
  }

#pragma unroll
  for (int nt = 0; nt < 2; ++nt) {
    const int col = colBase + nt * 16 + lo;
    const float bb = bias[col];
#pragma unroll
    for (int mt = 0; mt < 4; ++mt) {
#pragma unroll
      for (int v = 0; v < 8; ++v) {
        float val = acc[mt][nt][v] + bb;
        int row = rowBase + mt * 16 + v + 8 * hi;  // C layout: VGPR v -> M row
        if (which == 0) {
          Qo[row * ADIM + col] = f2bf(val);
        } else if (which == 1) {
          Ko[row * ADIM + col] = f2bf(val);
        } else {
          // Scatter V into per-block transposed layout Vt[g][dh][n2].
          int g  = row >> 7;
          int n2 = ((row & 127) << 3) + (col >> 6);
          int dh = col & 63;
          Vt[(g << 16) + (dh << 10) + n2] = f2bf(val);
        }
      }
    }
  }
}

// ----------------------- flash attention per block -------------------------
// grid = (8 q-tiles, 64 blocks). 8 waves x 16 query rows = 128 rows / WG.
__global__ __launch_bounds__(256) void attn_kernel(
    const unsigned short* __restrict__ Qb,   // [NG][1024][64] (== flat 8192x512)
    const unsigned short* __restrict__ Kb,   // same layout
    const unsigned short* __restrict__ Vt,   // [NG][64][1024]
    unsigned short* __restrict__ Ob) {       // [NG][1024][64]
  const int g  = blockIdx.y;
  const int q0 = blockIdx.x * 128;
  const int wave = threadIdx.x >> 5;
  const int lane = threadIdx.x & 31;
  const int lo = lane & 15, hi = lane >> 4;
  const int qbase = q0 + wave * 16;

  const unsigned short* Qg = Qb + (g << 16);
  const unsigned short* Kg = Kb + (g << 16);
  const unsigned short* Vg = Vt + (g << 16);

  // P relayout buffer: C-layout f32 -> A-layout bf16, 16x64 per wave.
  __shared__ __align__(16) unsigned short Pl[8][16 * 64];
  unsigned short* plw = &Pl[wave][0];

  // Q fragments for both 32-wide K steps of the 16x64 A tile (loaded once).
  FragB16 qf[2];
#pragma unroll
  for (int ks = 0; ks < 2; ++ks) {
    const unsigned short* qp = Qg + (qbase + lo) * DH + ks * 32;
    qf[ks].q[0] = *(const v4u*)(qp + hi * 16);
    qf[ks].q[1] = *(const v4u*)(qp + hi * 16 + 8);
  }

  v8f o[4];
  float m[8], l[8];
#pragma unroll
  for (int t = 0; t < 4; ++t)
#pragma unroll
    for (int e = 0; e < 8; ++e) o[t][e] = 0.0f;
#pragma unroll
  for (int v = 0; v < 8; ++v) { m[v] = -3.0e38f; l[v] = 0.0f; }

  for (int kc = 0; kc < GROWS; kc += 64) {
    if (kc + 64 < GROWS)  // hint next K chunk into cache (global_prefetch_b8)
      __builtin_prefetch(Kg + (kc + 64 + lane) * DH, 0, 1);

    // S(16x64) = Q(16x64) @ K^T, K-dim = dh = 64 (2 steps), N = 64 keys (4 tiles)
    v8f s[4];
#pragma unroll
    for (int nt = 0; nt < 4; ++nt)
#pragma unroll
      for (int e = 0; e < 8; ++e) s[nt][e] = 0.0f;

#pragma unroll
    for (int ks = 0; ks < 2; ++ks) {
#pragma unroll
      for (int nt = 0; nt < 4; ++nt) {
        // B column = key row of Kg (rows are contiguous dh vectors).
        const unsigned short* kp =
            Kg + (kc + nt * 16 + lo) * DH + ks * 32 + hi * 16;
        FragB16 kf;
        kf.q[0] = *(const v4u*)kp;
        kf.q[1] = *(const v4u*)(kp + 8);
        s[nt] = wmma_bf16(qf[ks], kf, s[nt]);
      }
    }
#pragma unroll
    for (int nt = 0; nt < 4; ++nt)
#pragma unroll
      for (int e = 0; e < 8; ++e) s[nt][e] *= SCALE;

    // Row max across this chunk (rows live at VGPR v, lanes span 16 cols/half).
    float tmax[8];
#pragma unroll
    for (int v = 0; v < 8; ++v) {
      float t = fmaxf(fmaxf(s[0][v], s[1][v]), fmaxf(s[2][v], s[3][v]));
#pragma unroll
      for (int mask = 1; mask < 16; mask <<= 1)
        t = fmaxf(t, __shfl_xor(t, mask, 16));
      tmax[v] = t;
    }

    // Online softmax rescale of running state.
#pragma unroll
    for (int v = 0; v < 8; ++v) {
      float mn = fmaxf(m[v], tmax[v]);
      float corr = __expf(m[v] - mn);
      m[v] = mn;
      l[v] *= corr;
#pragma unroll
      for (int t = 0; t < 4; ++t) o[t][v] *= corr;
    }

    // P = exp(S - m); accumulate row sums; stage bf16 P into LDS (A layout src).
    float psum[8];
#pragma unroll
    for (int v = 0; v < 8; ++v) psum[v] = 0.0f;
#pragma unroll
    for (int nt = 0; nt < 4; ++nt) {
#pragma unroll
      for (int v = 0; v < 8; ++v) {
        float p = __expf(s[nt][v] - m[v]);
        psum[v] += p;
        plw[(v + 8 * hi) * 64 + nt * 16 + lo] = f2bf(p);
      }
    }
#pragma unroll
    for (int v = 0; v < 8; ++v) {
      float t = psum[v];
#pragma unroll
      for (int mask = 1; mask < 16; mask <<= 1) t += __shfl_xor(t, mask, 16);
      l[v] += t;
    }

    // O(16x64) += P(16x64keys) @ Vchunk(64keys x 64dh)
#pragma unroll
    for (int ks = 0; ks < 2; ++ks) {
      FragB16 pa;  // read P back in A-fragment layout
      pa.q[0] = *(const v4u*)(plw + lo * 64 + ks * 32 + hi * 16);
      pa.q[1] = *(const v4u*)(plw + lo * 64 + ks * 32 + hi * 16 + 8);
#pragma unroll
      for (int nt = 0; nt < 4; ++nt) {
        // B column = dh; Vt rows (per dh) are contiguous over keys.
        const unsigned short* vp =
            Vg + (nt * 16 + lo) * GROWS + kc + ks * 32 + hi * 16;
        FragB16 vf;
        vf.q[0] = *(const v4u*)vp;
        vf.q[1] = *(const v4u*)(vp + 8);
        o[nt] = wmma_bf16(pa, vf, o[nt]);
      }
    }
  }

  // Normalize and store (flat layout == final O matrix 8192x512).
#pragma unroll
  for (int v = 0; v < 8; ++v) {
    float inv = 1.0f / l[v];
    int row = qbase + v + 8 * hi;
#pragma unroll
    for (int nt = 0; nt < 4; ++nt)
      Ob[(g << 16) + row * DH + nt * 16 + lo] = f2bf(o[nt][v] * inv);
  }
}

// --------------------------- output projection -----------------------------
// out(8192x768 f32) = O(8192x512 bf16) @ Wp + bp
__global__ __launch_bounds__(256) void out_gemm(
    const unsigned short* __restrict__ A,    // 8192x512 bf16
    const unsigned short* __restrict__ Wt,   // Wp^T: 768x512 bf16
    const float* __restrict__ bias,          // 768
    float* __restrict__ out) {               // 8192x768 f32
  const int m0   = blockIdx.y * 128;
  const int n0   = blockIdx.x * 128;
  const int wave = threadIdx.x >> 5;
  const int lane = threadIdx.x & 31;
  const int lo = lane & 15, hi = lane >> 4;
  const int rowBase = m0 + (wave >> 2) * 64;
  const int colBase = n0 + (wave & 3) * 32;

  v8f acc[4][2];
#pragma unroll
  for (int i = 0; i < 4; ++i)
#pragma unroll
    for (int j = 0; j < 2; ++j)
#pragma unroll
      for (int e = 0; e < 8; ++e) acc[i][j][e] = 0.0f;

  for (int k0 = 0; k0 < ADIM; k0 += 32) {
    FragB16 a[4], b[2];
#pragma unroll
    for (int mt = 0; mt < 4; ++mt) {
      const unsigned short* ap = A + (rowBase + mt * 16 + lo) * ADIM + k0;
      a[mt].q[0] = *(const v4u*)(ap + hi * 16);
      a[mt].q[1] = *(const v4u*)(ap + hi * 16 + 8);
    }
#pragma unroll
    for (int nt = 0; nt < 2; ++nt) {
      const unsigned short* bp =
          Wt + (colBase + nt * 16 + lo) * ADIM + k0 + hi * 16;
      b[nt].q[0] = *(const v4u*)bp;
      b[nt].q[1] = *(const v4u*)(bp + 8);
    }
#pragma unroll
    for (int mt = 0; mt < 4; ++mt)
#pragma unroll
      for (int nt = 0; nt < 2; ++nt)
        acc[mt][nt] = wmma_bf16(a[mt], b[nt], acc[mt][nt]);
  }

#pragma unroll
  for (int nt = 0; nt < 2; ++nt) {
    const int col = colBase + nt * 16 + lo;
    const float bb = bias[col];
#pragma unroll
    for (int mt = 0; mt < 4; ++mt)
#pragma unroll
      for (int v = 0; v < 8; ++v) {
        int row = rowBase + mt * 16 + v + 8 * hi;
        out[row * DMODEL + col] = acc[mt][nt][v] + bb;
      }
  }
}

// ------------------------------- launcher ----------------------------------

extern "C" void kernel_launch(void* const* d_in, const int* in_sizes, int n_in,
                              void* d_out, int out_size, void* d_ws,
                              size_t ws_size, hipStream_t stream) {
  (void)in_sizes; (void)n_in; (void)out_size; (void)ws_size;
  const float* x  = (const float*)d_in[0];
  const float* Wq = (const float*)d_in[1];
  const float* bq = (const float*)d_in[2];
  const float* Wk = (const float*)d_in[3];
  const float* bk = (const float*)d_in[4];
  const float* Wv = (const float*)d_in[5];
  const float* bv = (const float*)d_in[6];
  const float* Wp = (const float*)d_in[7];
  const float* bp = (const float*)d_in[8];
  float* out = (float*)d_out;

  char* ws = (char*)d_ws;
  size_t off = 0;
  auto alloc = [&](size_t bytes) -> char* {
    char* p = ws + off;
    off += (bytes + 255) & ~size_t(255);
    return p;
  };
  unsigned short* Xbf = (unsigned short*)alloc((size_t)SEQ * DMODEL * 2);
  unsigned short* Wqt = (unsigned short*)alloc((size_t)ADIM * DMODEL * 2);
  unsigned short* Wkt = (unsigned short*)alloc((size_t)ADIM * DMODEL * 2);
  unsigned short* Wvt = (unsigned short*)alloc((size_t)ADIM * DMODEL * 2);
  unsigned short* Wpt = (unsigned short*)alloc((size_t)DMODEL * ADIM * 2);
  unsigned short* Qbf = (unsigned short*)alloc((size_t)SEQ * ADIM * 2);
  unsigned short* Kbf = (unsigned short*)alloc((size_t)SEQ * ADIM * 2);
  unsigned short* Vtp = (unsigned short*)alloc((size_t)SEQ * ADIM * 2);
  unsigned short* Obf = (unsigned short*)alloc((size_t)SEQ * ADIM * 2);

  cvt_f32_to_bf16<<<(SEQ * DMODEL) / 256, 256, 0, stream>>>(x, Xbf,
                                                            SEQ * DMODEL);
  transpose_to_bf16<<<(DMODEL * ADIM) / 256, 256, 0, stream>>>(Wq, Wqt, DMODEL,
                                                               ADIM);
  transpose_to_bf16<<<(DMODEL * ADIM) / 256, 256, 0, stream>>>(Wk, Wkt, DMODEL,
                                                               ADIM);
  transpose_to_bf16<<<(DMODEL * ADIM) / 256, 256, 0, stream>>>(Wv, Wvt, DMODEL,
                                                               ADIM);
  transpose_to_bf16<<<(DMODEL * ADIM) / 256, 256, 0, stream>>>(Wp, Wpt, ADIM,
                                                               DMODEL);

  qkv_gemm<<<dim3(ADIM / 128, SEQ / 128, 3), 256, 0, stream>>>(
      Xbf, Wqt, Wkt, Wvt, bq, bk, bv, Qbf, Kbf, Vtp);

  attn_kernel<<<dim3(GROWS / 128, NG), 256, 0, stream>>>(Qbf, Kbf, Vtp, Obf);

  out_gemm<<<dim3(DMODEL / 128, SEQ / 128), 256, 0, stream>>>(Obf, Wpt, bp,
                                                              out);
}
